// CausalSelfAttentionRoPE_85899345920474
// MI455X (gfx1250) — matrix operand order repaired and checked by reading, not compiled, not run on hardware
//
#include <hip/hip_runtime.h>

typedef __attribute__((ext_vector_type(16))) _Float16 v16h;
typedef __attribute__((ext_vector_type(8)))  _Float16 v8h;
typedef __attribute__((ext_vector_type(8)))  float    v8f;

#define T_SEQ 2048
#define C_DIM 1024
#define HD    64
#define NH    16
#define BSZ   2
#define M_ROWS (BSZ * T_SEQ)      // 4096
#define N3    (3 * C_DIM)         // 3072

__device__ __forceinline__ v16h combine16(v8h lo, v8h hi) {
    v16h r;
#pragma unroll
    for (int i = 0; i < 8; ++i) { r[i] = lo[i]; r[i + 8] = hi[i]; }
    return r;
}

__device__ __forceinline__ v8f wmma_f16(v16h a, v16h b, v8f c) {
    return __builtin_amdgcn_wmma_f32_16x16x32_f16(false, a, false, b, (short)0, c,
                                                  false, false);
}

// ---------------- pre-pass: f32 -> f16 convert ----------------
__global__ void cvt_f16_kernel(const float* __restrict__ in,
                               _Float16* __restrict__ out, int n) {
    int i = (blockIdx.x * blockDim.x + threadIdx.x) * 8;
    if (i >= n) return;
    float4 a = *(const float4*)(in + i);
    float4 b = *(const float4*)(in + i + 4);
    v8h o;
    o[0] = (_Float16)a.x; o[1] = (_Float16)a.y; o[2] = (_Float16)a.z; o[3] = (_Float16)a.w;
    o[4] = (_Float16)b.x; o[5] = (_Float16)b.y; o[6] = (_Float16)b.z; o[7] = (_Float16)b.w;
    *(v8h*)(out + i) = o;
}

// W (K x N) f32 row-major -> Wt (N x K) f16 row-major
__global__ void tcvt_kernel(const float* __restrict__ W,
                            _Float16* __restrict__ Wt, int K, int N) {
    int idx = blockIdx.x * blockDim.x + threadIdx.x;
    if (idx >= N * K) return;
    int n = idx / K, k = idx - n * K;
    Wt[idx] = (_Float16)W[(size_t)k * N + n];
}

// ---------------- fused QKV GEMM + bias + RoPE (2x4 register blocking) -------
// Xh: (4096 x 1024) f16, Wt: (3072 x 1024) f16.
// One wave -> 32x64 macro-tile (8 WMMAs / k-step from 6 fragment loads).
__global__ __launch_bounds__(256) void gemm_qkv_rope(
    const _Float16* __restrict__ Xh, const _Float16* __restrict__ Wt,
    const float* __restrict__ bias, _Float16* __restrict__ Qo,
    _Float16* __restrict__ Ko, _Float16* __restrict__ Vto) {
    const int wave = threadIdx.x >> 5, lane = threadIdx.x & 31;
    const int l16 = lane & 15, hi16 = lane >> 4;
    const int tid = blockIdx.x * 8 + wave;           // 6144 macro tiles
    const int tm = tid & 127;                        // 128 macro M (32 rows)
    const int tn = tid >> 7;                         // 48 macro N (64 cols)

    const _Float16* arow0 = Xh + (size_t)(tm * 32 + l16) * C_DIM;
    const _Float16* arow1 = arow0 + (size_t)16 * C_DIM;
    const _Float16* brow[4];
#pragma unroll
    for (int j = 0; j < 4; ++j)
        brow[j] = Wt + (size_t)(tn * 64 + j * 16 + l16) * C_DIM;

    v8f acc[2][4];
#pragma unroll
    for (int i = 0; i < 2; ++i)
#pragma unroll
        for (int j = 0; j < 4; ++j) acc[i][j] = v8f{};

#pragma unroll 2
    for (int kb = 0; kb < C_DIM; kb += 32) {
        v16h a0 = combine16(*(const v8h*)(arow0 + kb + hi16 * 8),
                            *(const v8h*)(arow0 + kb + 16 + hi16 * 8));
        v16h a1 = combine16(*(const v8h*)(arow1 + kb + hi16 * 8),
                            *(const v8h*)(arow1 + kb + 16 + hi16 * 8));
        v16h bf[4];
#pragma unroll
        for (int j = 0; j < 4; ++j)
            bf[j] = combine16(*(const v8h*)(brow[j] + kb + hi16 * 8),
                              *(const v8h*)(brow[j] + kb + 16 + hi16 * 8));
#pragma unroll
        for (int j = 0; j < 4; ++j) {
            acc[0][j] = wmma_f16(a0, bf[j], acc[0][j]);
            acc[1][j] = wmma_f16(a1, bf[j], acc[1][j]);
        }
    }

#pragma unroll
    for (int i = 0; i < 2; ++i) {
#pragma unroll
        for (int j = 0; j < 4; ++j) {
            const int ng = (tn * 4 + j) * 16 + l16;
            const float bv = bias[ng];
            const int region = ng >> 10;             // 0=q,1=k,2=v
            const int np = ng & (C_DIM - 1);
            const int hh = np >> 6, d = np & (HD - 1);
            const int tmBase = (tm * 2 + i) * 16;

            if (region < 2) {
                const float fr =
                    __powf(10000.0f, -(float)(d & 31) * (1.0f / 32.0f));
                _Float16* dst = (region == 0) ? Qo : Ko;
#pragma unroll
                for (int r = 0; r < 8; ++r) {
                    const int mg = tmBase + r + 8 * hi16;
                    const int bb = mg >> 11, t = mg & (T_SEQ - 1);
                    float v   = acc[i][j][r] + bv;
                    float prt = __shfl_xor(v, 1, 32);    // pair partner channel
                    float rot = (d & 1) ? prt : -prt;
                    float sn, cs;
                    __sincosf((float)t * fr, &sn, &cs);
                    float o = v * cs + rot * sn;
                    dst[(((size_t)(bb * NH + hh) * T_SEQ + t) * HD) + d] =
                        (_Float16)o;
                }
            } else {
                const int bb = tmBase >> 11, tb = tmBase & (T_SEQ - 1);
                v8h hv;
#pragma unroll
                for (int r = 0; r < 8; ++r)
                    hv[r] = (_Float16)(acc[i][j][r] + bv);
                *(v8h*)(Vto + ((size_t)(bb * NH + hh) * HD + d) * T_SEQ + tb +
                        hi16 * 8) = hv;
            }
        }
    }
}

// ---------------- flash attention (one wave per 16-row q block per head) -----
#define P_STRIDE 40   // padded LDS row stride (halves), keeps 16B alignment
__global__ __launch_bounds__(128) void attn_kernel(
    const _Float16* __restrict__ Q, const _Float16* __restrict__ Kk,
    const _Float16* __restrict__ Vt, _Float16* __restrict__ Yh) {
    __shared__ __align__(16) _Float16 plds[4][16 * P_STRIDE];
    const int wave = threadIdx.x >> 5, lane = threadIdx.x & 31;
    const int l16 = lane & 15, hi16 = lane >> 4;
    const int qb = blockIdx.x * 4 + wave;            // 0..127
    const int qbase = qb * 16;
    const int bh = blockIdx.y;                       // 0..31
    const int bb = bh >> 4, h = bh & 15;

    const _Float16* Qb = Q  + (size_t)bh * T_SEQ * HD;
    const _Float16* Kb = Kk + (size_t)bh * T_SEQ * HD;
    const _Float16* Vb = Vt + (size_t)bh * HD * T_SEQ;
    _Float16* pl = plds[wave];

    // Q fragments (rows qbase..qbase+15, d split 0..31 / 32..63)
    const _Float16* qrow = Qb + (size_t)(qbase + l16) * HD;
    v16h qf0 = combine16(*(const v8h*)(qrow + hi16 * 8),
                         *(const v8h*)(qrow + 16 + hi16 * 8));
    v16h qf1 = combine16(*(const v8h*)(qrow + 32 + hi16 * 8),
                         *(const v8h*)(qrow + 48 + hi16 * 8));

    v8f accO[4] = {v8f{}, v8f{}, v8f{}, v8f{}};
    float mi[8], li[8];
#pragma unroll
    for (int r = 0; r < 8; ++r) { mi[r] = -1e30f; li[r] = 0.0f; }

    for (int k0 = 0; k0 < qbase + 16; k0 += 32) {
        v8f S[2];
#pragma unroll
        for (int s = 0; s < 2; ++s) {
            const int n0 = k0 + s * 16;
            const int tk = n0 + l16;
            const _Float16* krow = Kb + (size_t)tk * HD;
            v16h kf0 = combine16(*(const v8h*)(krow + hi16 * 8),
                                 *(const v8h*)(krow + 16 + hi16 * 8));
            v16h kf1 = combine16(*(const v8h*)(krow + 32 + hi16 * 8),
                                 *(const v8h*)(krow + 48 + hi16 * 8));
            v8f a = {};
            a = wmma_f16(qf0, kf0, a);
            a = wmma_f16(qf1, kf1, a);
#pragma unroll
            for (int r = 0; r < 8; ++r) {            // scale + causal mask
                float v = a[r] * 0.125f;
                const int tq = qbase + r + 8 * hi16;
                if (tk > tq) v = -1e30f;
                a[r] = v;
            }
            S[s] = a;
        }
        float mnew[8];
#pragma unroll
        for (int r = 0; r < 8; ++r) {                // row max across 16 lanes
            float mx = fmaxf(S[0][r], S[1][r]);
            mx = fmaxf(mx, __shfl_xor(mx, 1, 32));
            mx = fmaxf(mx, __shfl_xor(mx, 2, 32));
            mx = fmaxf(mx, __shfl_xor(mx, 4, 32));
            mx = fmaxf(mx, __shfl_xor(mx, 8, 32));
            mnew[r] = fmaxf(mi[r], mx);
        }
#pragma unroll
        for (int s = 0; s < 2; ++s)
#pragma unroll
            for (int r = 0; r < 8; ++r) {
                float p = __expf(S[s][r] - mnew[r]);
                S[s][r] = p;
                pl[(r + 8 * hi16) * P_STRIDE + s * 16 + l16] = (_Float16)p;
            }
#pragma unroll
        for (int r = 0; r < 8; ++r) {
            float rs = S[0][r] + S[1][r];
            rs += __shfl_xor(rs, 1, 32);
            rs += __shfl_xor(rs, 2, 32);
            rs += __shfl_xor(rs, 4, 32);
            rs += __shfl_xor(rs, 8, 32);
            float corr = __expf(mi[r] - mnew[r]);
            li[r] = li[r] * corr + rs;
            mi[r] = mnew[r];
#pragma unroll
            for (int dt = 0; dt < 4; ++dt) accO[dt][r] *= corr;
        }
        asm volatile("s_wait_dscnt 0x0" ::: "memory");   // P stores visible
        v16h pf = combine16(*(const v8h*)(pl + l16 * P_STRIDE + hi16 * 8),
                            *(const v8h*)(pl + l16 * P_STRIDE + 16 + hi16 * 8));
#pragma unroll
        for (int dt = 0; dt < 4; ++dt) {
            v16h vf = combine16(*(const v8h*)(Vb + (size_t)(dt * 16 + l16) * T_SEQ +
                                              k0 + hi16 * 8),
                                *(const v8h*)(Vb + (size_t)(dt * 16 + l16) * T_SEQ +
                                              k0 + 16 + hi16 * 8));
            accO[dt] = wmma_f16(pf, vf, accO[dt]);
        }
        // DS ops are in-order per wave: next iter's stores follow these loads.
    }
#pragma unroll
    for (int r = 0; r < 8; ++r) li[r] = 1.0f / li[r];
#pragma unroll
    for (int dt = 0; dt < 4; ++dt)
#pragma unroll
        for (int r = 0; r < 8; ++r) {
            const int tq = qbase + r + 8 * hi16;
            Yh[((size_t)bb * T_SEQ + tq) * C_DIM + h * HD + dt * 16 + l16] =
                (_Float16)(accO[dt][r] * li[r]);
        }
}

// ---------------- output projection GEMM + bias -> f32 (2x4 blocking) --------
__global__ __launch_bounds__(256) void gemm_out(
    const _Float16* __restrict__ Yh, const _Float16* __restrict__ Wt,
    const float* __restrict__ bias, float* __restrict__ Out) {
    const int wave = threadIdx.x >> 5, lane = threadIdx.x & 31;
    const int l16 = lane & 15, hi16 = lane >> 4;
    const int tid = blockIdx.x * 8 + wave;           // 2048 macro tiles
    const int tm = tid & 127;                        // 128 macro M
    const int tn = tid >> 7;                         // 16 macro N

    const _Float16* arow0 = Yh + (size_t)(tm * 32 + l16) * C_DIM;
    const _Float16* arow1 = arow0 + (size_t)16 * C_DIM;
    const _Float16* brow[4];
#pragma unroll
    for (int j = 0; j < 4; ++j)
        brow[j] = Wt + (size_t)(tn * 64 + j * 16 + l16) * C_DIM;

    v8f acc[2][4];
#pragma unroll
    for (int i = 0; i < 2; ++i)
#pragma unroll
        for (int j = 0; j < 4; ++j) acc[i][j] = v8f{};

#pragma unroll 2
    for (int kb = 0; kb < C_DIM; kb += 32) {
        v16h a0 = combine16(*(const v8h*)(arow0 + kb + hi16 * 8),
                            *(const v8h*)(arow0 + kb + 16 + hi16 * 8));
        v16h a1 = combine16(*(const v8h*)(arow1 + kb + hi16 * 8),
                            *(const v8h*)(arow1 + kb + 16 + hi16 * 8));
        v16h bf[4];
#pragma unroll
        for (int j = 0; j < 4; ++j)
            bf[j] = combine16(*(const v8h*)(brow[j] + kb + hi16 * 8),
                              *(const v8h*)(brow[j] + kb + 16 + hi16 * 8));
#pragma unroll
        for (int j = 0; j < 4; ++j) {
            acc[0][j] = wmma_f16(a0, bf[j], acc[0][j]);
            acc[1][j] = wmma_f16(a1, bf[j], acc[1][j]);
        }
    }

#pragma unroll
    for (int i = 0; i < 2; ++i)
#pragma unroll
        for (int j = 0; j < 4; ++j) {
            const int ng = (tn * 4 + j) * 16 + l16;
            const float bv = bias[ng];
#pragma unroll
            for (int r = 0; r < 8; ++r) {
                const int mg = (tm * 2 + i) * 16 + r + 8 * hi16;
                Out[(size_t)mg * C_DIM + ng] = acc[i][j][r] + bv;
            }
        }
}

extern "C" void kernel_launch(void* const* d_in, const int* in_sizes, int n_in,
                              void* d_out, int out_size, void* d_ws, size_t ws_size,
                              hipStream_t stream) {
    const float* x     = (const float*)d_in[0];
    const float* W_qkv = (const float*)d_in[1];
    const float* b_qkv = (const float*)d_in[2];
    const float* W_out = (const float*)d_in[3];
    const float* b_out = (const float*)d_in[4];
    float* out = (float*)d_out;

    char* ws = (char*)d_ws;
    _Float16* x_h    = (_Float16*)(ws + 0);                  //  8 MB
    _Float16* Wt_qkv = (_Float16*)(ws + (8u << 20));         //  6 MB
    _Float16* Wt_out = (_Float16*)(ws + (14u << 20));        //  2 MB
    _Float16* Qf     = (_Float16*)(ws + (16u << 20));        //  8 MB
    _Float16* Kf     = (_Float16*)(ws + (24u << 20));        //  8 MB
    _Float16* Vtf    = (_Float16*)(ws + (32u << 20));        //  8 MB
    _Float16* y_h    = (_Float16*)(ws + (40u << 20));        //  8 MB
    if (ws_size < (48u << 20)) return;

    // 1) precision pre-pass
    cvt_f16_kernel<<<(M_ROWS * C_DIM / 8 + 255) / 256, 256, 0, stream>>>(
        x, x_h, M_ROWS * C_DIM);
    tcvt_kernel<<<(N3 * C_DIM + 255) / 256, 256, 0, stream>>>(
        W_qkv, Wt_qkv, C_DIM, N3);
    tcvt_kernel<<<(C_DIM * C_DIM + 255) / 256, 256, 0, stream>>>(
        W_out, Wt_out, C_DIM, C_DIM);

    // 2) fused QKV projection + bias + RoPE (6144 macro tiles, 8 waves/block)
    gemm_qkv_rope<<<(M_ROWS / 32) * (N3 / 64) / 8, 256, 0, stream>>>(
        x_h, Wt_qkv, b_qkv, Qf, Kf, Vtf);

    // 3) causal flash attention: grid (T/16/4, B*H), 4 waves/block
    attn_kernel<<<dim3(T_SEQ / 16 / 4, BSZ * NH), 128, 0, stream>>>(
        Qf, Kf, Vtf, y_h);

    // 4) output projection (2048 macro tiles)
    gemm_out<<<(M_ROWS / 32) * (C_DIM / 64) / 8, 256, 0, stream>>>(
        y_h, Wt_out, b_out, out);
}
